// MambaEncoder_8409545966073
// MI455X (gfx1250) — matrix operand order repaired and checked by reading, compiled, not run on hardware
//
#include <hip/hip_runtime.h>
#include <hip/hip_bf16.h>
#include <math.h>

// ---------------------------------------------------------------------------
// Mamba encoder for MI455X (gfx1250, wave32, WMMA).
// GEMMs: v_wmma_f32_16x16x32_f16 with double-buffered LDS staging via
// global_load_async_to_lds_b128 (ASYNCcnt-tracked async copies).
// ---------------------------------------------------------------------------

#define Bsz 4
#define Tsz 1024
#define Dsz 512
#define D_INNER 1024
#define D_STATE 16
#define DT_RANK 32
#define FFN_DIM 2048
#define Mtok (Bsz * Tsz) // 4096 tokens

typedef __attribute__((ext_vector_type(16))) _Float16 v16h;
typedef __attribute__((ext_vector_type(8)))  _Float16 v8h;
typedef __attribute__((ext_vector_type(8)))  float    v8f;

// ---------------------------------------------------------------------------
// elementwise helpers
// ---------------------------------------------------------------------------
__device__ __forceinline__ float silu_f(float x) { return x / (1.0f + __expf(-x)); }

__device__ __forceinline__ float apply_act(float v, int act) {
  if (act == 1) {                       // softplus
    return (v > 20.0f) ? v : log1pf(expf(v));
  } else if (act == 2) {                // exact GELU
    return 0.5f * v * (1.0f + erff(v * 0.70710678118654752f));
  }
  return v;
}

// Async 16B global -> LDS copy (ASYNCcnt). lds_off = wave-relative LDS byte
// address (low 32 bits of the generic pointer to a __shared__ object).
__device__ __forceinline__ void async_load_b128(uint32_t lds_off, const void* gptr) {
  asm volatile("global_load_async_to_lds_b128 %0, %1, off"
               :: "v"(lds_off), "v"((unsigned long long)(uintptr_t)gptr)
               : "memory");
}
__device__ __forceinline__ void wait_asynccnt0() {
  asm volatile("s_wait_asynccnt 0x0" ::: "memory");
}

// ---------------------------------------------------------------------------
// x = x_in + sinusoidal positional encoding
// ---------------------------------------------------------------------------
__global__ void __launch_bounds__(256) k_pe_add(const float* __restrict__ xin,
                                                float* __restrict__ X) {
  int idx = blockIdx.x * 256 + threadIdx.x;
  if (idx >= Mtok * Dsz) return;
  int bt = idx >> 9;            // /512
  int d  = idx & (Dsz - 1);
  int t  = bt & (Tsz - 1);
  float freq = expf(-(float)(d & ~1) * (logf(10000.0f) / (float)Dsz));
  float ang  = (float)t * freq;
  float pe   = (d & 1) ? cosf(ang) : sinf(ang);
  X[idx] = xin[idx] + pe;
}

// ---------------------------------------------------------------------------
// LayerNorm over last dim (512). One row (token) per 256-thread block.
// Safe for src == dst.
// ---------------------------------------------------------------------------
__global__ void __launch_bounds__(256) k_layernorm(const float* __restrict__ src,
                                                   float* __restrict__ dst,
                                                   const float* __restrict__ g,
                                                   const float* __restrict__ b) {
  __shared__ float s_sum[8], s_sq[8];
  int row = blockIdx.x;
  int tid = threadIdx.x;
  const float* p = src + (size_t)row * Dsz;
  float v0 = p[tid];
  float v1 = p[tid + 256];
  float sum = v0 + v1;
  float sq  = v0 * v0 + v1 * v1;
  #pragma unroll
  for (int off = 16; off > 0; off >>= 1) {
    sum += __shfl_down(sum, off);
    sq  += __shfl_down(sq,  off);
  }
  if ((tid & 31) == 0) { s_sum[tid >> 5] = sum; s_sq[tid >> 5] = sq; }
  __syncthreads();
  float tsum = 0.0f, tsq = 0.0f;
  #pragma unroll
  for (int i = 0; i < 8; ++i) { tsum += s_sum[i]; tsq += s_sq[i]; }
  float mean = tsum * (1.0f / (float)Dsz);
  float var  = tsq  * (1.0f / (float)Dsz) - mean * mean;
  float rs   = rsqrtf(var + 1e-5f);
  float* q = dst + (size_t)row * Dsz;
  q[tid]       = (v0 - mean) * rs * g[tid]       + b[tid];
  q[tid + 256] = (v1 - mean) * rs * g[tid + 256] + b[tid + 256];
}

// ---------------------------------------------------------------------------
// f32 -> f16 (strided source rows), dst contiguous [rows, cols]
// ---------------------------------------------------------------------------
__global__ void __launch_bounds__(256) k_to_f16(const float* __restrict__ src, int src_ld,
                                                _Float16* __restrict__ dst,
                                                int rows, int cols) {
  int idx = blockIdx.x * 256 + threadIdx.x;
  if (idx >= rows * cols) return;
  int r = idx / cols;
  int c = idx - r * cols;
  dst[idx] = (_Float16)src[(size_t)r * src_ld + c];
}

// ---------------------------------------------------------------------------
// WMMA GEMM: Out[M,N] (=|+=) act( Xh[M,K] * Wh[N,K]^T + bias )
// Block = 256 threads = 8 waves (4x2); wave tile 32x32 (2x2 WMMA tiles).
// Block tile 128x64; K staged 32 at a time through double-buffered LDS using
// async global->LDS b128 copies. Requires M%128==0, N%64==0, K%32==0.
// ---------------------------------------------------------------------------
#define LDT 40                     // padded LDS row stride in halves (80B = 20 banks)

__global__ void __launch_bounds__(256) k_wmma_gemm(const _Float16* __restrict__ Xh,
                                                   const _Float16* __restrict__ Wh,
                                                   const float* __restrict__ bias,
                                                   float* __restrict__ Out,
                                                   int M, int N, int K,
                                                   int act, int resadd) {
  __shared__ __align__(16) _Float16 shA[2][128 * LDT];
  __shared__ __align__(16) _Float16 shB[2][64 * LDT];

  const int tid  = threadIdx.x;
  const int lane = tid & 31;
  const int wave = tid >> 5;
  const int wm = wave >> 1;             // 0..3
  const int wn = wave & 1;              // 0..1
  const int mblk = blockIdx.y * 128;
  const int nblk = blockIdx.x * 64;
  const int half = lane >> 4;           // 0 or 1
  const int l16  = lane & 15;

  // Async-copy work assignment: A tile 128x32 halves = 512 x 16B chunks
  // (2 per thread), B tile 64x32 halves = 256 chunks (1 per thread).
  const int ar0 = (tid * 2) >> 2,     ac0 = (tid * 2) & 3;
  const int ar1 = (tid * 2 + 1) >> 2, ac1 = (tid * 2 + 1) & 3;
  const int br  = tid >> 2,           bc  = tid & 3;

  auto prefetch = [&](int buf, int kb) {
    async_load_b128((uint32_t)(uintptr_t)&shA[buf][ar0 * LDT + ac0 * 8],
                    Xh + (size_t)(mblk + ar0) * K + kb + ac0 * 8);
    async_load_b128((uint32_t)(uintptr_t)&shA[buf][ar1 * LDT + ac1 * 8],
                    Xh + (size_t)(mblk + ar1) * K + kb + ac1 * 8);
    async_load_b128((uint32_t)(uintptr_t)&shB[buf][br * LDT + bc * 8],
                    Wh + (size_t)(nblk + br) * K + kb + bc * 8);
  };

  v8f acc[2][2] = {};

  prefetch(0, 0);
  for (int kb = 0; kb < K; kb += 32) {
    const int cur = (kb >> 5) & 1;
    wait_asynccnt0();                 // this wave's copies into shX[cur] landed
    __syncthreads();                  // all waves' copies visible; prev buffer fully consumed
    if (kb + 32 < K) prefetch(cur ^ 1, kb + 32);

    // CDNA5 wave32 WMMA operand layouts (ISA 7.12.2):
    //  A 16x32 f16: lane = row l16; K = 8*half+{0..7} and 16+8*half+{0..7}
    //  B 32x16 f16: lane = col l16; K = 8*half+{0..7} and 16+8*half+{0..7}
    v16h a[2], b[2];
    #pragma unroll
    for (int mi = 0; mi < 2; ++mi) {
      const _Float16* p = &shA[cur][(wm * 32 + mi * 16 + l16) * LDT + 8 * half];
      v8h lo = *(const v8h*)p;
      v8h hi = *(const v8h*)(p + 16);
      a[mi] = __builtin_shufflevector(lo, hi, 0,1,2,3,4,5,6,7,8,9,10,11,12,13,14,15);
    }
    #pragma unroll
    for (int ni = 0; ni < 2; ++ni) {
      const _Float16* p = &shB[cur][(wn * 32 + ni * 16 + l16) * LDT + 8 * half];
      v8h lo = *(const v8h*)p;
      v8h hi = *(const v8h*)(p + 16);
      b[ni] = __builtin_shufflevector(lo, hi, 0,1,2,3,4,5,6,7,8,9,10,11,12,13,14,15);
    }
    #pragma unroll
    for (int mi = 0; mi < 2; ++mi)
      #pragma unroll
      for (int ni = 0; ni < 2; ++ni)
        acc[mi][ni] = __builtin_amdgcn_wmma_f32_16x16x32_f16(
            false, a[mi], false, b[ni], (short)0, acc[mi][ni], false, false);
  }

  // Writeback: C/D layout — VGPR r holds (M = r + 8*half, N = l16)
  #pragma unroll
  for (int mi = 0; mi < 2; ++mi) {
    #pragma unroll
    for (int ni = 0; ni < 2; ++ni) {
      int n = nblk + wn * 32 + ni * 16 + l16;
      float bv = bias ? bias[n] : 0.0f;
      #pragma unroll
      for (int r = 0; r < 8; ++r) {
        int m = mblk + wm * 32 + mi * 16 + half * 8 + r;
        float v = apply_act(acc[mi][ni][r] + bv, act);
        size_t o = (size_t)m * N + n;
        if (resadd) Out[o] += v; else Out[o] = v;
      }
    }
  }
}

// ---------------------------------------------------------------------------
// Depthwise causal conv (K=4) over time + bias + SiLU.
// Input: XZ [Mtok, 2048] (uses cols 0..1023). Output: XI [Mtok, 1024].
// ---------------------------------------------------------------------------
__global__ void __launch_bounds__(256) k_conv_silu(const float* __restrict__ xz,
                                                   const float* __restrict__ w,  // [1024,4]
                                                   const float* __restrict__ bias,
                                                   float* __restrict__ out) {
  int idx = blockIdx.x * 256 + threadIdx.x;
  if (idx >= Mtok * D_INNER) return;
  int bt = idx >> 10;
  int c  = idx & (D_INNER - 1);
  int t  = bt & (Tsz - 1);
  int b  = bt >> 10;
  float acc = bias[c];
  #pragma unroll
  for (int j = 0; j < 4; ++j) {
    int tt = t + j - 3;
    if (tt >= 0)
      acc += xz[(size_t)(b * Tsz + tt) * (2 * D_INNER) + c] * w[c * 4 + j];
  }
  out[idx] = silu_f(acc);
}

// ---------------------------------------------------------------------------
// Selective scan. One thread per (batch, inner-channel): 4096 threads, each
// carries h[16] in registers across T=1024 steps.
// ---------------------------------------------------------------------------
__global__ void __launch_bounds__(256) k_scan(const float* __restrict__ delta, // [Mtok,1024]
                                              const float* __restrict__ xi,    // [Mtok,1024]
                                              const float* __restrict__ xdbl,  // [Mtok,64]
                                              const float* __restrict__ A_log, // [1024,16]
                                              const float* __restrict__ Dp,    // [1024]
                                              float* __restrict__ Yout) {      // [Mtok,1024]
  int tid = blockIdx.x * 256 + threadIdx.x;   // 0..4095
  int b = tid >> 10;
  int d = tid & (D_INNER - 1);
  float A[D_STATE], h[D_STATE];
  #pragma unroll
  for (int n = 0; n < D_STATE; ++n) { A[n] = -expf(A_log[d * D_STATE + n]); h[n] = 0.0f; }
  float Dv = Dp[d];
  for (int t = 0; t < Tsz; ++t) {
    int bt = b * Tsz + t;
    float dl = delta[(size_t)bt * D_INNER + d];
    float xv = xi[(size_t)bt * D_INNER + d];
    const float* xr = xdbl + (size_t)bt * 64;
    float y = 0.0f;
    #pragma unroll
    for (int n = 0; n < D_STATE; ++n) {
      float dA = expf(dl * A[n]);
      h[n] = dA * h[n] + dl * xr[DT_RANK + n] * xv;
      y += h[n] * xr[DT_RANK + D_STATE + n];
    }
    Yout[(size_t)bt * D_INNER + d] = y + xv * Dv;
  }
}

// y * silu(z) -> f16 (z = XZ[:, 1024:2048])
__global__ void __launch_bounds__(256) k_gate_f16(const float* __restrict__ Y,
                                                  const float* __restrict__ xz,
                                                  _Float16* __restrict__ H) {
  int idx = blockIdx.x * 256 + threadIdx.x;
  if (idx >= Mtok * D_INNER) return;
  int row = idx >> 10;
  int col = idx & (D_INNER - 1);
  float z = xz[(size_t)row * (2 * D_INNER) + D_INNER + col];
  H[idx] = (_Float16)(Y[idx] * silu_f(z));
}

__global__ void __launch_bounds__(256) k_mask(float* __restrict__ X,
                                              const int* __restrict__ lengths) {
  int idx = blockIdx.x * 256 + threadIdx.x;
  if (idx >= Mtok * Dsz) return;
  int bt = idx >> 9;
  int t = bt & (Tsz - 1);
  int b = bt >> 10;
  if (t >= lengths[b]) X[idx] = 0.0f;
}

__global__ void __launch_bounds__(256) k_copy(const float* __restrict__ src,
                                              float* __restrict__ dst) {
  int idx = blockIdx.x * 256 + threadIdx.x;
  if (idx < Mtok * Dsz) dst[idx] = src[idx];
}

// ---------------------------------------------------------------------------
// Host orchestration
// ---------------------------------------------------------------------------
static inline dim3 grid1(size_t n) { return dim3((unsigned)((n + 255) / 256)); }

static void launch_gemm(const _Float16* Xh, const _Float16* Wh, const float* bias,
                        float* Out, int M, int N, int K, int act, int resadd,
                        hipStream_t s) {
  dim3 g(N / 64, M / 128);
  k_wmma_gemm<<<g, 256, 0, s>>>(Xh, Wh, bias, Out, M, N, K, act, resadd);
}

extern "C" void kernel_launch(void* const* d_in, const int* in_sizes, int n_in,
                              void* d_out, int out_size, void* d_ws, size_t ws_size,
                              hipStream_t stream) {
  (void)in_sizes; (void)n_in; (void)out_size; (void)ws_size;

  const float* x_in    = (const float*)d_in[0];
  const int*   lengths = (const int*)d_in[1];

  // Workspace layout (fp32 unless noted) — everything L2-resident on MI455X.
  float* X     = (float*)d_ws;                    // [4096,512]   residual
  float* LNOUT = X     + (size_t)Mtok * Dsz;      // [4096,512]
  float* XZ    = LNOUT + (size_t)Mtok * Dsz;      // [4096,2048]  in_proj out / FFN hidden
  float* XI    = XZ    + (size_t)Mtok * 2048;     // [4096,1024]  conv+silu
  float* XDBL  = XI    + (size_t)Mtok * 1024;     // [4096,64]
  float* DELTA = XDBL  + (size_t)Mtok * 64;       // [4096,1024]
  float* Yb    = DELTA + (size_t)Mtok * 1024;     // [4096,1024]
  _Float16* H16 = (_Float16*)(Yb + (size_t)Mtok * 1024); // [4096,2048] f16 act
  _Float16* W16 = H16 + (size_t)Mtok * 2048;             // f16 weights (<=1M elems)

  k_pe_add<<<grid1((size_t)Mtok * Dsz), 256, 0, stream>>>(x_in, X);

  // Param flattening: JAX dict sorted-key order per layer:
  // A_log, D, conv_b, conv_w, dt_proj_b, dt_proj_w, ffn_b1, ffn_b2, ffn_w1,
  // ffn_w2, in_proj_w, [ln0_b, ln0_g,] ln1_b, ln1_g, ln2_b, ln2_g, out_proj_w, x_proj_w
  const int base[2] = {2, 21};
  for (int L = 0; L < 2; ++L) {
    int bs = base[L];
    const float* A_log  = (const float*)d_in[bs + 0];
    const float* Dp     = (const float*)d_in[bs + 1];
    const float* conv_b = (const float*)d_in[bs + 2];
    const float* conv_w = (const float*)d_in[bs + 3];
    const float* dt_b   = (const float*)d_in[bs + 4];
    const float* dt_w   = (const float*)d_in[bs + 5];
    const float* ffn_b1 = (const float*)d_in[bs + 6];
    const float* ffn_b2 = (const float*)d_in[bs + 7];
    const float* ffn_w1 = (const float*)d_in[bs + 8];
    const float* ffn_w2 = (const float*)d_in[bs + 9];
    const float* inp_w  = (const float*)d_in[bs + 10];
    int lo = (L == 0) ? 2 : 0;
    const float* ln1_b  = (const float*)d_in[bs + 11 + lo];
    const float* ln1_g  = (const float*)d_in[bs + 12 + lo];
    const float* ln2_b  = (const float*)d_in[bs + 13 + lo];
    const float* ln2_g  = (const float*)d_in[bs + 14 + lo];
    const float* out_w  = (const float*)d_in[bs + 15 + lo];
    const float* xp_w   = (const float*)d_in[bs + 16 + lo];

    if (L == 0) {
      const float* ln0_b = (const float*)d_in[bs + 11];
      const float* ln0_g = (const float*)d_in[bs + 12];
      k_layernorm<<<Mtok, 256, 0, stream>>>(X, X, ln0_g, ln0_b);
    }

    // ---- Mamba block ----
    k_layernorm<<<Mtok, 256, 0, stream>>>(X, LNOUT, ln1_g, ln1_b);
    k_to_f16<<<grid1((size_t)Mtok * Dsz), 256, 0, stream>>>(LNOUT, Dsz, H16, Mtok, Dsz);
    k_to_f16<<<grid1((size_t)2048 * 512), 256, 0, stream>>>(inp_w, 512, W16, 2048, 512);
    launch_gemm(H16, W16, nullptr, XZ, Mtok, 2048, 512, 0, 0, stream);     // in_proj

    k_conv_silu<<<grid1((size_t)Mtok * D_INNER), 256, 0, stream>>>(XZ, conv_w, conv_b, XI);

    k_to_f16<<<grid1((size_t)Mtok * D_INNER), 256, 0, stream>>>(XI, D_INNER, H16, Mtok, D_INNER);
    k_to_f16<<<grid1((size_t)64 * 1024), 256, 0, stream>>>(xp_w, 1024, W16, 64, 1024);
    launch_gemm(H16, W16, nullptr, XDBL, Mtok, 64, 1024, 0, 0, stream);    // x_proj

    k_to_f16<<<grid1((size_t)Mtok * DT_RANK), 256, 0, stream>>>(XDBL, 64, H16, Mtok, DT_RANK);
    k_to_f16<<<grid1((size_t)1024 * 32), 256, 0, stream>>>(dt_w, 32, W16, 1024, 32);
    launch_gemm(H16, W16, dt_b, DELTA, Mtok, 1024, 32, /*softplus*/1, 0, stream); // dt_proj

    k_scan<<<16, 256, 0, stream>>>(DELTA, XI, XDBL, A_log, Dp, Yb);

    k_gate_f16<<<grid1((size_t)Mtok * D_INNER), 256, 0, stream>>>(Yb, XZ, H16);
    k_to_f16<<<grid1((size_t)512 * 1024), 256, 0, stream>>>(out_w, 1024, W16, 512, 1024);
    launch_gemm(H16, W16, nullptr, X, Mtok, 512, 1024, 0, /*res+=*/1, stream);  // out_proj

    // ---- FFN block ----
    k_layernorm<<<Mtok, 256, 0, stream>>>(X, LNOUT, ln2_g, ln2_b);
    k_to_f16<<<grid1((size_t)Mtok * Dsz), 256, 0, stream>>>(LNOUT, Dsz, H16, Mtok, Dsz);
    k_to_f16<<<grid1((size_t)2048 * 512), 256, 0, stream>>>(ffn_w1, 512, W16, 2048, 512);
    launch_gemm(H16, W16, ffn_b1, XZ, Mtok, 2048, 512, /*gelu*/2, 0, stream);   // ffn1

    k_to_f16<<<grid1((size_t)Mtok * 2048), 256, 0, stream>>>(XZ, 2048, H16, Mtok, 2048);
    k_to_f16<<<grid1((size_t)512 * 2048), 256, 0, stream>>>(ffn_w2, 2048, W16, 512, 2048);
    launch_gemm(H16, W16, ffn_b2, X, Mtok, 512, 2048, 0, /*res+=*/1, stream);   // ffn2

    k_mask<<<grid1((size_t)Mtok * Dsz), 256, 0, stream>>>(X, lengths);
  }

  k_copy<<<grid1((size_t)Mtok * Dsz), 256, 0, stream>>>(X, (float*)d_out);
}
